// CRFLoss_55946243997950
// MI455X (gfx1250) — compile-verified, matrix operand order checked by reading, not run on hardware
//
#include <hip/hip_runtime.h>
#include <hip/hip_bf16.h>
#include <math.h>

typedef __attribute__((ext_vector_type(16))) _Float16 v16h;
typedef __attribute__((ext_vector_type(8)))  float    v8f;

#define BB 512
#define SS 1024
#define TT 64
#define BM 16        // batch rows per workgroup (WMMA M)
#define NWAVE 4
#define NTHREADS 128
#define PSTR 72      // p_lds row stride in halves (padded)

// Raw CDNA5 transcendentals (TRANS32: co-execute with VALU).
// Args here are always positive normals / bounded, so no guard sequences needed.
__device__ __forceinline__ float fast_exp(float x) {
    return __builtin_amdgcn_exp2f(x * 1.4426950408889634f);
}
__device__ __forceinline__ float fast_log(float x) {
    return __builtin_amdgcn_logf(x) * 0.6931471805599453f;
}

// ---------------------------------------------------------------------------
// Phase 0: gold-path score per batch:
//   total[b] = sum_s mask[b,s] * ( emit[b,s,tags[b,s]] + (s>0 ? trans[tags[b,s-1],tags[b,s]] : 0) )
// ---------------------------------------------------------------------------
__global__ __launch_bounds__(NTHREADS)
void crf_total_score(const int* __restrict__ tags,
                     const unsigned char* __restrict__ mask,
                     const float* __restrict__ emit,
                     const float* __restrict__ trans,
                     float* __restrict__ total)
{
    const int b   = blockIdx.x;
    const int tid = threadIdx.x;
    const int*           tg = tags + (size_t)b * SS;
    const unsigned char* mk = mask + (size_t)b * SS;
    const float*         em = emit + (size_t)b * SS * TT;

    float acc = 0.0f;
    for (int s = tid; s < SS; s += NTHREADS) {
        if (mk[s]) {
            int t  = tg[s];
            float v = em[(size_t)s * TT + t];
            if (s > 0) v += trans[tg[s - 1] * TT + t];
            acc += v;
        }
    }
    // wave32 reduce, then cross-wave via LDS
    #pragma unroll
    for (int off = 16; off > 0; off >>= 1) acc += __shfl_down(acc, off, 32);
    __shared__ float red[NWAVE];
    if ((tid & 31) == 0) red[tid >> 5] = acc;
    __syncthreads();
    if (tid == 0) {
        float s0 = 0.0f;
        for (int w = 0; w < NWAVE; ++w) s0 += red[w];
        total[b] = s0;
    }
}

// ---------------------------------------------------------------------------
// Phase 1: forward scan.  Per step:
//   M[m]   = max_k d[m,k]
//   p[m,k] = exp(d[m,k]-M[m])                       (f16)
//   sum    = P (16x64) x expT (64x64)               (WMMA f16 -> f32 acc)
//   cand   = M[m] + log(sum) + emit[b,s,n]
//   d      = mask ? cand : d
// Each wave owns one 16-wide N tile; K=64 -> 2 chained wmma_f32_16x16x32_f16.
// ---------------------------------------------------------------------------
__global__ __launch_bounds__(NTHREADS)
void crf_forward_scan(const unsigned char* __restrict__ mask,
                      const float* __restrict__ emit,
                      const float* __restrict__ trans,
                      const float* __restrict__ total,
                      float* __restrict__ out)
{
    __shared__ float          d_lds[BM * TT];      // forward variables (f32)
    __shared__ _Float16       p_lds[BM * PSTR];    // exp(d - rowmax) (f16)
    __shared__ float          rowMax[BM];
    __shared__ unsigned short mask_words[64];      // bit r = mask[row r] at step s0+c

    const int tid   = threadIdx.x;
    const int lane  = tid & 31;
    const int wave  = tid >> 5;
    const int b0    = blockIdx.x * BM;
    const int n0    = wave * 16;                   // this wave's N tile
    const int khalf = (lane >= 16) ? 8 : 0;        // K sub-offset of lane half
    const int nl    = lane & 15;

    // --- B fragments: expT columns [n0, n0+16), K chunks 0..31 / 32..63.
    //     16-bit B layout: lane = N (mod 16); element e -> K = 16*(e>>3) + khalf + (e&7).
    v16h bfrag0, bfrag1;
    #pragma unroll
    for (int e = 0; e < 16; ++e) {
        int k = ((e >> 3) << 4) + khalf + (e & 7);
        bfrag0[e] = (_Float16)fast_exp(trans[k        * TT + n0 + nl]);
        bfrag1[e] = (_Float16)fast_exp(trans[(k + 32) * TT + n0 + nl]);
    }

    // --- d0 = emit[:, 0, :]
    for (int i = tid; i < BM * TT; i += NTHREADS) {
        int r = i >> 6, cidx = i & 63;
        d_lds[i] = emit[((size_t)(b0 + r) * SS) * TT + cidx];
    }
    // mask chunk covering s in [0, 63], packed: bit r of word c = mask[b0+r][c]
    if (tid < 64) {
        unsigned int w = 0;
        #pragma unroll
        for (int r = 0; r < BM; ++r)
            w |= (mask[(size_t)(b0 + r) * SS + tid] ? 1u : 0u) << r;
        mask_words[tid] = (unsigned short)w;
    }
    __syncthreads();

    // row-stat thread mapping: 8 threads per row, 8 cols each
    const int srow = tid >> 3;
    const int scol = (tid & 7) * 8;

    // initial rowMax / p
    {
        float mx = -INFINITY, vals[8];
        #pragma unroll
        for (int j = 0; j < 8; ++j) { vals[j] = d_lds[srow * TT + scol + j]; mx = fmaxf(mx, vals[j]); }
        #pragma unroll
        for (int m2 = 1; m2 < 8; m2 <<= 1) mx = fmaxf(mx, __shfl_xor(mx, m2, 32));
        if ((tid & 7) == 0) rowMax[srow] = mx;
        #pragma unroll
        for (int j = 0; j < 8; ++j) p_lds[srow * PSTR + scol + j] = (_Float16)fast_exp(vals[j] - mx);
    }
    __syncthreads();

    for (int s = 1; s < SS; ++s) {
        if ((s & 63) == 0) {                       // refill packed mask chunk
            __syncthreads();
            if (tid < 64) {
                unsigned int w = 0;
                #pragma unroll
                for (int r = 0; r < BM; ++r)
                    w |= (mask[(size_t)(b0 + r) * SS + s + tid] ? 1u : 0u) << r;
                mask_words[tid] = (unsigned short)w;
            }
            __syncthreads();
        }
        // prefetch emit 8 steps ahead (global_prefetch_b8)
        if (tid < 32 && s + 8 < SS) {
            int r = tid >> 1, h = (tid & 1) * 32;
            __builtin_prefetch(&emit[((size_t)(b0 + r) * SS + s + 8) * TT + h], 0, 1);
        }

        // A fragments from p_lds (16-bit A layout: lane = M mod 16)
        v16h a0, a1;
        {
            const _Float16* pr = &p_lds[nl * PSTR];
            #pragma unroll
            for (int e = 0; e < 16; ++e) {
                int k = ((e >> 3) << 4) + khalf + (e & 7);
                a0[e] = pr[k];
                a1[e] = pr[k + 32];
            }
        }
        v8f acc = {};
        acc = __builtin_amdgcn_wmma_f32_16x16x32_f16(false, a0, false, bfrag0, (short)0, acc, false, false);
        acc = __builtin_amdgcn_wmma_f32_16x16x32_f16(false, a1, false, bfrag1, (short)0, acc, false, false);

        // single ds_load_u16: 16 mask bits for this step
        const unsigned int mw = mask_words[s & 63] >> khalf;

        // C layout: vgpr r, lanes 0-15 -> M=r, lanes 16-31 -> M=r+8; N = n0 + (lane&15)
        float dnew[8];
        #pragma unroll
        for (int r = 0; r < 8; ++r) {
            int m = r + khalf;
            int n = n0 + nl;
            float dold = d_lds[m * TT + n];
            float ev   = emit[((size_t)(b0 + m) * SS + s) * TT + n];
            float cand = rowMax[m] + fast_log(acc[r]) + ev;
            dnew[r] = ((mw >> r) & 1u) ? cand : dold;
        }
        __syncthreads();                           // everyone done reading old d
        #pragma unroll
        for (int r = 0; r < 8; ++r)
            d_lds[(r + khalf) * TT + n0 + nl] = dnew[r];
        __syncthreads();                           // new d visible

        // rowMax / p for next step
        {
            float mx = -INFINITY, vals[8];
            #pragma unroll
            for (int j = 0; j < 8; ++j) { vals[j] = d_lds[srow * TT + scol + j]; mx = fmaxf(mx, vals[j]); }
            #pragma unroll
            for (int m2 = 1; m2 < 8; m2 <<= 1) mx = fmaxf(mx, __shfl_xor(mx, m2, 32));
            if ((tid & 7) == 0) rowMax[srow] = mx;
            #pragma unroll
            for (int j = 0; j < 8; ++j) p_lds[srow * PSTR + scol + j] = (_Float16)fast_exp(vals[j] - mx);
        }
        __syncthreads();                           // p/rowMax visible before next A-frag read
    }

    // final: log_z[b] = rowMax + log(sum exp(d - rowMax));  out = log_z - total
    {
        float mx  = rowMax[srow];
        float sum = 0.0f;
        #pragma unroll
        for (int j = 0; j < 8; ++j) sum += fast_exp(d_lds[srow * TT + scol + j] - mx);
        #pragma unroll
        for (int m2 = 1; m2 < 8; m2 <<= 1) sum += __shfl_xor(sum, m2, 32);
        if ((tid & 7) == 0) {
            float logz = mx + fast_log(sum);
            out[b0 + srow] = logz - total[b0 + srow];
        }
    }
}

// ---------------------------------------------------------------------------
extern "C" void kernel_launch(void* const* d_in, const int* in_sizes, int n_in,
                              void* d_out, int out_size, void* d_ws, size_t ws_size,
                              hipStream_t stream) {
    const int*           tags  = (const int*)d_in[0];
    const unsigned char* mask  = (const unsigned char*)d_in[1];   // jax bool -> 1 byte
    const float*         emit  = (const float*)d_in[2];
    const float*         trans = (const float*)d_in[3];
    float* out   = (float*)d_out;
    float* total = (float*)d_ws;                                  // 512 floats of scratch

    crf_total_score<<<dim3(BB), dim3(NTHREADS), 0, stream>>>(tags, mask, emit, trans, total);
    crf_forward_scan<<<dim3(BB / BM), dim3(NTHREADS), 0, stream>>>(mask, emit, trans, total, out);
}
